// MyModel_75239237091436
// MI455X (gfx1250) — matrix operand order repaired and checked by reading, not compiled, not run on hardware
//
#include <hip/hip_runtime.h>
#include <cstddef>
#include <cstdint>

#define NN      16384      // N_NODES
#define NE      262144     // N_EDGES
#define ADJ     64
#define IN_F    128
#define HID     128
#define HEADS   4
#define SLOPEF  0.2f

typedef float    v2f  __attribute__((ext_vector_type(2)));
typedef float    v8f  __attribute__((ext_vector_type(8)));
typedef _Float16 v16h __attribute__((ext_vector_type(16)));

__device__ __forceinline__ float eluf(float x) { return x > 0.f ? x : (__expf(x) - 1.f); }

// order-preserving float <-> uint encoding for atomicMax-based segment max
__device__ __forceinline__ unsigned encf(float f) {
    unsigned u = __float_as_uint(f);
    return (u & 0x80000000u) ? ~u : (u | 0x80000000u);
}
__device__ __forceinline__ float decf(unsigned e) {
    unsigned u = (e & 0x80000000u) ? (e & 0x7fffffffu) : ~e;
    return __uint_as_float(u);
}

// ----------------------------- fills ---------------------------------------
__global__ void fill_f32_k(float* p, float v, int n) {
    int i = blockIdx.x * blockDim.x + threadIdx.x;
    if (i < n) p[i] = v;
}
__global__ void fill_u32_k(unsigned* p, unsigned v, int n) {
    int i = blockIdx.x * blockDim.x + threadIdx.x;
    if (i < n) p[i] = v;
}
// out[i] = bias[i % N]  (split-K GEMM accumulates on top of this)
__global__ void bias_init_k(float* out, const float* bias, int n, int N) {
    int i = blockIdx.x * blockDim.x + threadIdx.x;
    if (i < n) out[i] = bias[i % N];
}

// ------------------------ WMMA fp32 GEMM ------------------------------------
// C[M,N] = act(A[M,K]) @ B[K,N] (+ bias[N]).  Row-major, lda=K, ldb=ldc=N.
// Block = 8 wave32s = 128 rows x 64 cols of C; all waves share one 64-wide
// column group, so B chunks (KC x 64 fp32) are staged through LDS once per
// block. Double-buffered staging: next chunk's global loads are in flight
// while the current chunk's 32 WMMAs consume LDS. LDS rows padded to 80
// floats (stride%64==16) so the two 16-lane halves of each ds read hit
// disjoint bank sets on the 64-bank LDS.
// SPLITK=1: blockIdx.y selects a K-slice of length kLen; partial products are
// accumulated into C (pre-initialized with the bias) via global_atomic_add_f32.
// Requires M%128==0, N%64==0, kLen%32==0.
#define KC      32
#define BSH_LD  80
template <int ACT, int SPLITK>
__global__ void gemm_wmma(const float* __restrict__ A, const float* __restrict__ B,
                          const float* __restrict__ bias, float* __restrict__ C,
                          int M, int N, int K, int kLen)
{
    __shared__ float Bsh[2][KC * BSH_LD];       // 2 x 10 KB of the 320 KB WGP LDS

    const int lane = threadIdx.x & 31;
    const int wave = threadIdx.x >> 5;
    const int nColGroups = N >> 6;              // 64 cols per block
    const int mb = blockIdx.x / nColGroups;     // 128-row block
    const int cg = blockIdx.x % nColGroups;
    const int m0 = (mb << 7) + (wave << 4);     // this wave's 16 rows
    const int n0 = cg << 6;

    const int kBegin = SPLITK ? (int)blockIdx.y * kLen : 0;
    const int nCh = (SPLITK ? kLen : K) / KC;

    const int rowA = m0 + (lane & 15);
    const int colB = lane & 15;
    const int hi   = lane >> 4;                 // 0: lanes 0-15, 1: lanes 16-31
    const int kSel = hi << 1;                   // A/B frag K-select: 0 or 2

    // cooperative B-stage mapping: 256 threads x 8 floats = 32x64 chunk
    const int ldRow = threadIdx.x >> 3;         // 0..31
    const int ldCol = (threadIdx.x & 7) << 3;   // 0,8,...,56

    v8f acc0 = {}, acc1 = {}, acc2 = {}, acc3 = {};

    // prologue: fetch + stage chunk 0
    float4 r0, r1;
    {
        const float* gp = B + (size_t)(kBegin + ldRow) * N + n0 + ldCol;
        r0 = *(const float4*)(gp);
        r1 = *(const float4*)(gp + 4);
        float* sp = &Bsh[0][ldRow * BSH_LD + ldCol];
        *(float4*)(sp)     = r0;
        *(float4*)(sp + 4) = r1;
    }
    __syncthreads();

    for (int ch = 0; ch < nCh; ++ch) {
        const int kc = kBegin + ch * KC;
        // issue next chunk's global loads early (latency overlaps compute)
        if (ch + 1 < nCh) {
            const float* gp = B + (size_t)(kc + KC + ldRow) * N + n0 + ldCol;
            r0 = *(const float4*)(gp);
            r1 = *(const float4*)(gp + 4);
        }
        const float* __restrict__ Bcur = Bsh[ch & 1];

#if __has_builtin(__builtin_amdgcn_wmma_f32_16x16x4_f32)
        // Full-precision path: V_WMMA_F32_16X16X4_F32.
        // A frag 16x4: lanes 0-15 hold K=k0+0/1, lanes 16-31 hold K=k0+2/3.
        // B frag 4x16: lanes 0-15 hold K=k0+0/1 of col, lanes 16-31 K=k0+2/3.
        #pragma unroll
        for (int kk = 0; kk < KC; kk += 4) {
            float x0 = A[(size_t)rowA * K + kc + kk + kSel];
            float x1 = A[(size_t)rowA * K + kc + kk + kSel + 1];
            if (ACT) { x0 = eluf(x0); x1 = eluf(x1); }
            v2f a; a.x = x0; a.y = x1;
            const float* bp = &Bcur[(kk + kSel) * BSH_LD + colB];
            v2f b0, b1, b2, b3;
            b0.x = bp[0];  b0.y = bp[BSH_LD];
            b1.x = bp[16]; b1.y = bp[BSH_LD + 16];
            b2.x = bp[32]; b2.y = bp[BSH_LD + 32];
            b3.x = bp[48]; b3.y = bp[BSH_LD + 48];
            acc0 = __builtin_amdgcn_wmma_f32_16x16x4_f32(false, a, false, b0, (short)0, acc0, false, false);
            acc1 = __builtin_amdgcn_wmma_f32_16x16x4_f32(false, a, false, b1, (short)0, acc1, false, false);
            acc2 = __builtin_amdgcn_wmma_f32_16x16x4_f32(false, a, false, b2, (short)0, acc2, false, false);
            acc3 = __builtin_amdgcn_wmma_f32_16x16x4_f32(false, a, false, b3, (short)0, acc3, false, false);
        }
#else
        // Fallback: convert-on-load f16 inputs, f32 accumulate (codegen-confirmed builtin).
        {
            const int aBase = hi << 4;  // A: lanes<16 K=0..15; lanes>=16 K=16..31
            const int bBase = hi << 4;  // B: lanes<16 K=0..15; lanes>=16 K=16..31
            v16h a;
            #pragma unroll
            for (int j = 0; j < 16; ++j) {
                int koff = aBase + j;
                float x = A[(size_t)rowA * K + kc + koff];
                if (ACT) x = eluf(x);
                a[j] = (_Float16)x;
            }
            v16h b0, b1, b2, b3;
            #pragma unroll
            for (int j = 0; j < 16; ++j) {
                const float* bp = &Bcur[(bBase + j) * BSH_LD + colB];
                b0[j] = (_Float16)bp[0];
                b1[j] = (_Float16)bp[16];
                b2[j] = (_Float16)bp[32];
                b3[j] = (_Float16)bp[48];
            }
            acc0 = __builtin_amdgcn_wmma_f32_16x16x32_f16(false, a, false, b0, (short)0, acc0, false, false);
            acc1 = __builtin_amdgcn_wmma_f32_16x16x32_f16(false, a, false, b1, (short)0, acc1, false, false);
            acc2 = __builtin_amdgcn_wmma_f32_16x16x32_f16(false, a, false, b2, (short)0, acc2, false, false);
            acc3 = __builtin_amdgcn_wmma_f32_16x16x32_f16(false, a, false, b3, (short)0, acc3, false, false);
        }
#endif
        __syncthreads();                    // all waves done reading Bsh[ch&1]
        if (ch + 1 < nCh) {
            float* sp = &Bsh[(ch + 1) & 1][ldRow * BSH_LD + ldCol];
            *(float4*)(sp)     = r0;
            *(float4*)(sp + 4) = r1;
            __syncthreads();                // next buffer visible to all waves
        }
    }

    // C/D layout: VGPR v -> row (hi?8:0)+v, col = lane&15
    const int rBase = m0 + (hi << 3);
    const int c = lane & 15;
    if (SPLITK) {
        #pragma unroll
        for (int v = 0; v < 8; ++v) {
            float* Cp = C + (size_t)(rBase + v) * N + n0 + c;
            atomicAdd(&Cp[0],  acc0[v]);
            atomicAdd(&Cp[16], acc1[v]);
            atomicAdd(&Cp[32], acc2[v]);
            atomicAdd(&Cp[48], acc3[v]);
        }
    } else {
        const float bv0 = bias[n0 + c];
        const float bv1 = bias[n0 + 16 + c];
        const float bv2 = bias[n0 + 32 + c];
        const float bv3 = bias[n0 + 48 + c];
        #pragma unroll
        for (int v = 0; v < 8; ++v) {
            float* Cp = C + (size_t)(rBase + v) * N + n0 + c;
            Cp[0]  = acc0[v] + bv0;
            Cp[16] = acc1[v] + bv1;
            Cp[32] = acc2[v] + bv2;
            Cp[48] = acc3[v] + bv3;
        }
    }
}

// --------------------- edge phase 1: scores + segment max -------------------
// One wave per edge. score[e,h] = sum_f attn[h,f]*leakyrelu(fs[src]+fd[dst]).
__global__ void edge_scores_k(const float* __restrict__ fs, const float* __restrict__ fd,
                              const int* __restrict__ src, const int* __restrict__ dst,
                              const float* __restrict__ attn, float* __restrict__ score,
                              unsigned* __restrict__ mEnc, int E, int H, int F, float slope)
{
    int e = (blockIdx.x * blockDim.x + threadIdx.x) >> 5;
    int lane = threadIdx.x & 31;
    if (e >= E) return;
    int s = src[e], d = dst[e];
    for (int h = 0; h < H; ++h) {
        const float* ps = fs + ((size_t)s * H + h) * F;
        const float* pd = fd + ((size_t)d * H + h) * F;
        const float* pa = attn + (size_t)h * F;
        float acc = 0.f;
        for (int f = lane * 4; f < F; f += 128) {          // F==128 -> 1 iter, float4 coalesced
            float4 a4 = *(const float4*)(ps + f);
            float4 b4 = *(const float4*)(pd + f);
            float4 w4 = *(const float4*)(pa + f);
            float e0 = a4.x + b4.x, e1 = a4.y + b4.y, e2 = a4.z + b4.z, e3 = a4.w + b4.w;
            e0 = e0 > 0.f ? e0 : slope * e0;
            e1 = e1 > 0.f ? e1 : slope * e1;
            e2 = e2 > 0.f ? e2 : slope * e2;
            e3 = e3 > 0.f ? e3 : slope * e3;
            acc += e0 * w4.x + e1 * w4.y + e2 * w4.z + e3 * w4.w;
        }
        for (int off = 16; off; off >>= 1) acc += __shfl_xor(acc, off, 32);
        if (lane == 0) {
            score[(size_t)e * H + h] = acc;
            atomicMax(&mEnc[(size_t)d * H + h], encf(acc));
        }
    }
}

// --------------- edge phase 2: p = exp(score - max), denom sum --------------
__global__ void edge_norm_k(float* __restrict__ score, const unsigned* __restrict__ mEnc,
                            float* __restrict__ denom, const int* __restrict__ dst,
                            int E, int H)
{
    int i = blockIdx.x * blockDim.x + threadIdx.x;
    if (i >= E * H) return;
    int e = i / H, h = i - e * H;
    int d = dst[e];
    float p = __expf(score[i] - decf(mEnc[(size_t)d * H + h]));
    score[i] = p;                                          // overwrite in place with p
    atomicAdd(&denom[(size_t)d * H + h], p);
}

// --------------- edge phase 3: out[dst] += fs[src] * p/denom ----------------
// One wave per edge; float4 gather of fs, native f32 scatter-add into L2.
__global__ void edge_aggregate_k(const float* __restrict__ fs, const float* __restrict__ p,
                                 const float* __restrict__ denom,
                                 const int* __restrict__ src, const int* __restrict__ dst,
                                 float* __restrict__ out, int E, int H, int F)
{
    int e = (blockIdx.x * blockDim.x + threadIdx.x) >> 5;
    int lane = threadIdx.x & 31;
    if (e >= E) return;
    int s = src[e], d = dst[e];
    for (int h = 0; h < H; ++h) {
        float a = p[(size_t)e * H + h] / denom[(size_t)d * H + h];
        const float* ps = fs + ((size_t)s * H + h) * F;
        float* po = out + ((size_t)d * H + h) * F;
        for (int f = lane * 4; f < F; f += 128) {          // F==128 -> 1 iter
            float4 v4 = *(const float4*)(ps + f);
            atomicAdd(&po[f + 0], v4.x * a);
            atomicAdd(&po[f + 1], v4.y * a);
            atomicAdd(&po[f + 2], v4.z * a);
            atomicAdd(&po[f + 3], v4.w * a);
        }
    }
}

// ---------------------------------------------------------------------------
static inline int cdiv(int a, int b) { return (a + b - 1) / b; }

extern "C" void kernel_launch(void* const* d_in, const int* in_sizes, int n_in,
                              void* d_out, int out_size, void* d_ws, size_t ws_size,
                              hipStream_t stream)
{
    (void)in_sizes; (void)n_in; (void)out_size; (void)ws_size;
    const float* feat  = (const float*)d_in[0];
    const int*   src   = (const int*)d_in[1];
    const int*   dst   = (const int*)d_in[2];
    const float* W1s   = (const float*)d_in[3];
    const float* b1s   = (const float*)d_in[4];
    const float* W1d   = (const float*)d_in[5];
    const float* b1d   = (const float*)d_in[6];
    const float* attn1 = (const float*)d_in[7];
    const float* W2s   = (const float*)d_in[8];
    const float* b2s   = (const float*)d_in[9];
    const float* W2d   = (const float*)d_in[10];
    const float* b2d   = (const float*)d_in[11];
    const float* attn2 = (const float*)d_in[12];
    const float* Wn    = (const float*)d_in[13];
    const float* bn    = (const float*)d_in[14];
    float* out = (float*)d_out;

    // workspace layout (~114 MB total). Layer-2 buffers alias the layer-1
    // fs1/fd1 region, which is dead once layer-1 aggregation completes.
    char* w = (char*)d_ws;
    const size_t MB = (size_t)1 << 20;
    float*    fs1 = (float*)(w);                       // 32 MB [NN, 4*128]
    float*    fd1 = (float*)(w + 32 * MB);             // 32 MB
    float*    h1  = (float*)(w + 64 * MB);             // 32 MB
    float*    h2  = (float*)(w + 96 * MB);             //  8 MB [NN, 128]
    float*    sc1 = (float*)(w + 104 * MB);            //  4 MB [E, 4]
    unsigned* m1  = (unsigned*)(w + 108 * MB);         // 256 KB [NN, 4]
    float*    dn1 = (float*)(w + 108 * MB + 256 * 1024); // 256 KB
    float*    fs2 = (float*)(w);                       //  8 MB (alias)
    float*    fd2 = (float*)(w + 8 * MB);              //  8 MB (alias)
    float*    sc2 = (float*)(w + 16 * MB);             //  1 MB (alias)
    unsigned* m2  = (unsigned*)(w + 17 * MB);          // 64 KB (alias)
    float*    dn2 = (float*)(w + 17 * MB + 64 * 1024); // 64 KB (alias)

    const int B = 256;                    // 8 wave32s per block
    const unsigned ENC_NEG_INF = 0x007FFFFFu;   // encf(-inf)

    // ================= layer 1 =================
    fill_u32_k<<<cdiv(NN * HEADS, B), B, 0, stream>>>(m1, ENC_NEG_INF, NN * HEADS);
    fill_f32_k<<<cdiv(NN * HEADS, B), B, 0, stream>>>(dn1, 0.f, NN * HEADS);
    fill_f32_k<<<cdiv(NN * HEADS * HID, B), B, 0, stream>>>(h1, 0.f, NN * HEADS * HID);

    int g1 = (NN / 128) * ((HEADS * HID) / 64);          // blocks: 128x64 C-tiles
    gemm_wmma<0, 0><<<g1, B, 0, stream>>>(feat, W1s, b1s, fs1, NN, HEADS * HID, IN_F, IN_F);
    gemm_wmma<0, 0><<<g1, B, 0, stream>>>(feat, W1d, b1d, fd1, NN, HEADS * HID, IN_F, IN_F);

    edge_scores_k<<<NE / 8, B, 0, stream>>>(fs1, fd1, src, dst, attn1, sc1, m1, NE, HEADS, HID, SLOPEF);
    edge_norm_k<<<cdiv(NE * HEADS, B), B, 0, stream>>>(sc1, m1, dn1, dst, NE, HEADS);
    edge_aggregate_k<<<NE / 8, B, 0, stream>>>(fs1, sc1, dn1, src, dst, h1, NE, HEADS, HID);

    // ================= layer 2 (ELU fused into GEMM A-load) =================
    fill_u32_k<<<cdiv(NN, B), B, 0, stream>>>(m2, ENC_NEG_INF, NN);
    fill_f32_k<<<cdiv(NN, B), B, 0, stream>>>(dn2, 0.f, NN);
    fill_f32_k<<<cdiv(NN * HID, B), B, 0, stream>>>(h2, 0.f, NN * HID);

    int g2 = (NN / 128) * (HID / 64);
    gemm_wmma<1, 0><<<g2, B, 0, stream>>>(h1, W2s, b2s, fs2, NN, HID, HEADS * HID, HEADS * HID);
    gemm_wmma<1, 0><<<g2, B, 0, stream>>>(h1, W2d, b2d, fd2, NN, HID, HEADS * HID, HEADS * HID);

    edge_scores_k<<<NE / 8, B, 0, stream>>>(fs2, fd2, src, dst, attn2, sc2, m2, NE, 1, HID, SLOPEF);
    edge_norm_k<<<cdiv(NE, B), B, 0, stream>>>(sc2, m2, dn2, dst, NE, 1);
    edge_aggregate_k<<<NE / 8, B, 0, stream>>>(fs2, sc2, dn2, src, dst, h2, NE, 1, HID);

    // ===== head: elu(h2).reshape(256, 8192) @ Wn + bn, split-K over 32 =====
    const int bat = NN / ADJ;                       // 256
    const int KH = HID * ADJ;                       // 8192
    const int SLICES = 32;                          // kLen = 256 -> 8 chunks/slice
    bias_init_k<<<cdiv(bat * 128, B), B, 0, stream>>>(out, bn, bat * 128, 128);
    dim3 g3((bat / 128) * (128 / 64), SLICES);      // 4 x 32 = 128 blocks
    gemm_wmma<1, 1><<<g3, B, 0, stream>>>(h2, Wn, nullptr, out, bat, 128, KH, KH / SLICES);
}
